// GRU_4870492913947
// MI455X (gfx1250) — matrix operand order repaired and checked by reading, not compiled, not run on hardware
//
#include <hip/hip_runtime.h>

typedef __attribute__((ext_vector_type(16))) __bf16 v16bf;
typedef __attribute__((ext_vector_type(8)))  float  v8f;
typedef unsigned short u16;
typedef unsigned int   u32;
typedef unsigned long long u64;

#define BB 64
#define SS 512
#define II 512
#define HH 1024
#define NBLK 32
#define WPB  8   // waves per block

// Padded LDS row strides (elements): +8 so column-lane stride hits distinct banks.
#define LDW_IH 520    // 512 + 8
#define LDW_HH 1032   // 1024 + 8
#define LDS_ELEMS (96 * LDW_IH + 96 * LDW_HH)
#define LDS_BYTES (LDS_ELEMS * 2)

// sched_group_barrier masks
#define SG_WMMA 0x008
#define SG_VMEM_RD 0x020
#define SG_DS_RD 0x100

union FragBF {
  v16bf v;
  uint4 q[2];
};

static __device__ __forceinline__ v8f wmma_bf16(v16bf a, v16bf b, v8f c) {
  // D = A(16x32 bf16) x B(32x16 bf16) + C(16x16 f32)
  return __builtin_amdgcn_wmma_f32_16x16x32_bf16(false, a, false, b, (short)0, c, false, false);
}

// B fragment: 16 consecutive bf16 (two 16B LDS loads).
static __device__ __forceinline__ FragBF lds_frag(const u16* base, u32 off) {
  FragBF f;
  f.q[0] = *(const uint4*)(base + off);
  f.q[1] = *(const uint4*)(base + off + 8);
  return f;
}
// A fragment: 16 consecutive bf16 [k..k+15] (two 16B global loads).
static __device__ __forceinline__ FragBF glb_frag(const u16* base, u32 off) {
  FragBF f;
  f.q[0] = *(const uint4*)(base + off);
  f.q[1] = *(const uint4*)(base + off + 8);
  return f;
}

static __device__ __forceinline__ float sigmoid_f(float x) {
  return 1.0f / (1.0f + __expf(-x));
}
static __device__ __forceinline__ float tanh_f(float x) {
  float e = __expf(2.0f * x);
  return (e - 1.0f) / (e + 1.0f);
}

// Grid-wide barrier: all NBLK blocks resident (32 blocks x 256 thr, 1 block/WGP).
static __device__ __forceinline__ void grid_barrier(u32* cnt, u32* gen, u32 nblk) {
  __threadfence();
  __syncthreads();
  if (threadIdx.x == 0 && threadIdx.y == 0) {
    volatile u32* vgen = gen;
    u32 g = *vgen;
    if (atomicAdd(cnt, 1u) == nblk - 1u) {
      *cnt = 0u;
      __threadfence();
      atomicAdd(gen, 1u);
    } else {
      while (*vgen == g) { __builtin_amdgcn_s_sleep(1); }
    }
  }
  __syncthreads();
}

__global__ void f32_to_bf16_kernel(const float* __restrict__ src,
                                   u16* __restrict__ dst, int n) {
  int i = blockIdx.x * 256 + threadIdx.x;
  if (i < n) {
    union { __bf16 b; u16 u; } cv;
    cv.b = (__bf16)src[i];
    dst[i] = cv.u;
  }
}

__global__ void gru_init_kernel(float* __restrict__ h0f,
                                u16* __restrict__ h0b,
                                u32* __restrict__ sync) {
  int i = blockIdx.x * 256 + threadIdx.x;
  if (i < BB * HH) { h0f[i] = 0.0f; h0b[i] = (u16)0; }
  if (i < 2) sync[i] = 0u;
}

__global__ __launch_bounds__(256) void gru_scan_kernel(
    const u16* __restrict__ xbf,   // [B,S,I] bf16
    const u16* __restrict__ wih,   // [3H,I]  bf16
    const u16* __restrict__ whh,   // [3H,H]  bf16
    const float* __restrict__ bih, // [3H]
    const float* __restrict__ bhh, // [3H]
    float* __restrict__ hbuf,      // [2][B][H] f32 (ping-pong)
    u16*  __restrict__ hbf,        // [2][B][H] bf16 (ping-pong)
    u32*  __restrict__ sync,       // [2] barrier state
    float* __restrict__ out)       // [B,S,H] f32
{
  extern __shared__ __align__(16) u16 smem[];
  u16* lwih = smem;                     // [96][LDW_IH]  rows: (g*2+ntL)*16 + r
  u16* lwhh = smem + 96 * LDW_IH;       // [96][LDW_HH]

  const int lane = threadIdx.x;         // 0..31
  const int ty   = threadIdx.y;         // 0..7
  const int bk   = blockIdx.x;          // 0..31
  const int m    = ty >> 1;             // batch tile 0..3 (all m in one block)
  const int ntL  = ty & 1;              // local column tile 0..1
  const int nt   = bk * 2 + ntL;        // global column tile 0..63
  const int b0   = m * 16;
  const int c16  = lane & 15;
  const int col  = nt * 16 + c16;       // column in [0,H)
  const int khA  = (lane >> 4) * 16;    // A-frag per-lane K base
  const int khB  = (lane >> 4) * 16;    // B-frag per-lane K base
  const int tid  = ty * 32 + lane;      // 0..255

  // ---- Preload this block's weight slice into LDS (once; reused 512 steps).
  for (int i = tid; i < 96 * 64; i += 256) {
    const int rr = i >> 6;
    const int cc = (i & 63) * 8;
    const int t  = rr >> 4;
    const int g  = t >> 1, nl = t & 1;
    const int grow = g * HH + (bk * 2 + nl) * 16 + (rr & 15);
    *(uint4*)(lwih + rr * LDW_IH + cc) = *(const uint4*)(wih + (u64)grow * II + cc);
  }
  for (int i = tid; i < 96 * 128; i += 256) {
    const int rr = i >> 7;
    const int cc = (i & 127) * 8;
    const int t  = rr >> 4;
    const int g  = t >> 1, nl = t & 1;
    const int grow = g * HH + (bk * 2 + nl) * 16 + (rr & 15);
    *(uint4*)(lwhh + rr * LDW_HH + cc) = *(const uint4*)(whh + (u64)grow * HH + cc);
  }
  __syncthreads();

  // Per-lane biases (column-dependent only; same for all 8 acc regs).
  const float br  = bih[col]          + bhh[col];
  const float bz  = bih[HH + col]     + bhh[HH + col];
  const float bin = bih[2 * HH + col];
  const float bhn = bhh[2 * HH + col];

  // Per-lane LDS B-fragment bases (lane holds column N = c16 -> weight row).
  const u32 ibR = (u32)((0 * 2 + ntL) * 16 + c16) * LDW_IH + khB;
  const u32 ibZ = (u32)((1 * 2 + ntL) * 16 + c16) * LDW_IH + khB;
  const u32 ibN = (u32)((2 * 2 + ntL) * 16 + c16) * LDW_IH + khB;
  const u32 hbR = (u32)((0 * 2 + ntL) * 16 + c16) * LDW_HH + khB;
  const u32 hbZ = (u32)((1 * 2 + ntL) * 16 + c16) * LDW_HH + khB;
  const u32 hbN = (u32)((2 * 2 + ntL) * 16 + c16) * LDW_HH + khB;

  // A-operand row for this lane (32-bit element offsets from uniform bases).
  const int arow    = b0 + c16;
  const u32 xRowOff = (u32)arow * (SS * II) + khA;   // + s*II per step
  const u32 hRowOff = (u32)arow * HH + khA;

  for (int s = 0; s < SS; ++s) {
    const u32 pin  = (u32)(s & 1) * (BB * HH);
    const u32 pout = (u32)((s + 1) & 1) * (BB * HH);
    const u32 xo   = xRowOff + (u32)s * II;
    const u32 ho   = pin + hRowOff;

    // ---- Early issues: h_prev fp32 for the blend (ready; barrier-ordered).
    float hp[8];
    #pragma unroll
    for (int j = 0; j < 8; ++j) {
      const int brow = b0 + j + ((lane >> 4) << 3);
      hp[j] = hbuf[pin + (u32)brow * HH + col];
    }
    // Warm the cache path for next step's x row (speculative prefetch).
    {
      const int sn = (s + 1 < SS) ? (s + 1) : s;
      __builtin_prefetch((const void*)(xbf + xRowOff + (u32)sn * II), 0, 2);
    }

    v8f accR  = {br, br, br, br, br, br, br, br};
    v8f accZ  = {bz, bz, bz, bz, bz, bz, bz, bz};
    v8f accIN = {bin, bin, bin, bin, bin, bin, bin, bin};
    v8f accHN = {bhn, bhn, bhn, bhn, bhn, bhn, bhn, bhn};

    // ---- input projection: K = I = 512.  A global, B LDS.
    // Pipeline via sched groups: 2 iterations of loads run ahead of the WMMAs.
    {
      __builtin_amdgcn_sched_group_barrier(SG_VMEM_RD, 4, 0);   // A: iters 0,1
      __builtin_amdgcn_sched_group_barrier(SG_DS_RD, 12, 0);    // B: iters 0,1
      #pragma unroll
      for (int k0 = 0; k0 < II; k0 += 32) {
        FragBF a  = glb_frag(xbf, xo + k0);
        FragBF bR = lds_frag(lwih, ibR + k0);
        FragBF bZ = lds_frag(lwih, ibZ + k0);
        FragBF bN = lds_frag(lwih, ibN + k0);
        accR  = wmma_bf16(a.v, bR.v, accR);
        accZ  = wmma_bf16(a.v, bZ.v, accZ);
        accIN = wmma_bf16(a.v, bN.v, accIN);
        __builtin_amdgcn_sched_group_barrier(SG_WMMA, 3, 0);    // wmma k
        __builtin_amdgcn_sched_group_barrier(SG_VMEM_RD, 2, 0); // A k+2
        __builtin_amdgcn_sched_group_barrier(SG_DS_RD, 6, 0);   // B k+2
      }
    }

    // ---- hidden projection: K = H = 1024.  A global, B LDS.
    {
      __builtin_amdgcn_sched_group_barrier(SG_VMEM_RD, 4, 0);
      __builtin_amdgcn_sched_group_barrier(SG_DS_RD, 12, 0);
      #pragma unroll
      for (int k0 = 0; k0 < HH; k0 += 32) {
        FragBF a  = glb_frag(hbf, ho + k0);
        FragBF bR = lds_frag(lwhh, hbR + k0);
        FragBF bZ = lds_frag(lwhh, hbZ + k0);
        FragBF bN = lds_frag(lwhh, hbN + k0);
        accR  = wmma_bf16(a.v, bR.v, accR);
        accZ  = wmma_bf16(a.v, bZ.v, accZ);
        accHN = wmma_bf16(a.v, bN.v, accHN);
        __builtin_amdgcn_sched_group_barrier(SG_WMMA, 3, 0);
        __builtin_amdgcn_sched_group_barrier(SG_VMEM_RD, 2, 0);
        __builtin_amdgcn_sched_group_barrier(SG_DS_RD, 6, 0);
      }
    }

    // ---- gates + blend. D layout: reg j -> row j + 8*(lane>=16), col = lane%16
    #pragma unroll
    for (int j = 0; j < 8; ++j) {
      const int brow = b0 + j + ((lane >> 4) << 3);
      const float vr = sigmoid_f(accR[j]);
      const float vz = sigmoid_f(accZ[j]);
      const float vn = tanh_f(accIN[j] + vr * accHN[j]);
      const float hn = (1.0f - vz) * vn + vz * hp[j];
      out[(u32)brow * (SS * HH) + (u32)s * HH + col] = hn;
      hbuf[pout + (u32)brow * HH + col] = hn;
      union { __bf16 b; u16 u; } cv;
      cv.b = (__bf16)hn;
      hbf[pout + (u32)brow * HH + col] = cv.u;
    }

    grid_barrier(sync, sync + 1, NBLK);
  }
}

extern "C" void kernel_launch(void* const* d_in, const int* in_sizes, int n_in,
                              void* d_out, int out_size, void* d_ws, size_t ws_size,
                              hipStream_t stream) {
  (void)in_sizes; (void)n_in; (void)out_size; (void)ws_size;
  const float* x   = (const float*)d_in[0];   // [B,S,I]
  const float* wih = (const float*)d_in[1];   // [3H,I]
  const float* whh = (const float*)d_in[2];   // [3H,H]
  const float* bih = (const float*)d_in[3];   // [3H]
  const float* bhh = (const float*)d_in[4];   // [3H]
  float* out = (float*)d_out;                 // [B,S,H]

  char* ws = (char*)d_ws;
  size_t off = 0;
  auto carve = [&](size_t bytes) -> char* {
    char* p = ws + off;
    off = (off + bytes + 255) & ~(size_t)255;
    return p;
  };
  u16*  xbf  = (u16*)carve((size_t)BB * SS * II * sizeof(u16));     // 32 MB
  u16*  wihb = (u16*)carve((size_t)3 * HH * II * sizeof(u16));      // 3 MB
  u16*  whhb = (u16*)carve((size_t)3 * HH * HH * sizeof(u16));      // 6 MB
  float* hbuf = (float*)carve((size_t)2 * BB * HH * sizeof(float)); // 512 KB
  u16*  hbf  = (u16*)carve((size_t)2 * BB * HH * sizeof(u16));      // 256 KB
  u32*  sync = (u32*)carve(256);

  const int nx   = BB * SS * II;
  const int nwih = 3 * HH * II;
  const int nwhh = 3 * HH * HH;
  f32_to_bf16_kernel<<<(nx   + 255) / 256, 256, 0, stream>>>(x,   xbf,  nx);
  f32_to_bf16_kernel<<<(nwih + 255) / 256, 256, 0, stream>>>(wih, wihb, nwih);
  f32_to_bf16_kernel<<<(nwhh + 255) / 256, 256, 0, stream>>>(whh, whhb, nwhh);
  gru_init_kernel<<<(BB * HH + 255) / 256, 256, 0, stream>>>(hbuf, hbf, sync);

  // Opt in to >64KB dynamic LDS (gfx1250 WGP has 320KB).
  (void)hipFuncSetAttribute((const void*)gru_scan_kernel,
                            hipFuncAttributeMaxDynamicSharedMemorySize, LDS_BYTES);

  gru_scan_kernel<<<dim3(NBLK), dim3(32, WPB), LDS_BYTES, stream>>>(
      xbf, wihb, whhb, bih, bhh, hbuf, hbf, sync, out);
}
